// QLSTM_65481071407219
// MI455X (gfx1250) — matrix operand order repaired and checked by reading, not compiled, not run on hardware
//
#include <hip/hip_runtime.h>

#define T_DIM 512
#define B_DIM 512
#define D_DIM 512
#define H_DIM 512
#define GH    32
#define IN1   1024       // D + H
#define BT    16         // batch rows per workgroup
#define NWG   (B_DIM/BT) // 32 workgroups

typedef __attribute__((ext_vector_type(16))) __bf16 v16bf;
typedef __attribute__((ext_vector_type(8)))  __bf16 v8bf;
typedef __attribute__((ext_vector_type(4)))  __bf16 v4bf;
typedef __attribute__((ext_vector_type(8)))  float  v8f;
typedef __attribute__((ext_vector_type(4)))  float  v4f;
typedef __attribute__((ext_vector_type(4)))  unsigned int v4u;

__device__ __forceinline__ float sigmoid_fast(float x) {
    return 1.0f / (1.0f + __expf(-x));
}
__device__ __forceinline__ float tanh_fast(float x) {
    return 2.0f / (1.0f + __expf(-2.0f * x)) - 1.0f;
}

// ---- prep: cast + transpose weights into WMMA-B-friendly [g][n][k] bf16 ----
__global__ void qlstm_prep(const float* __restrict__ W1, const float* __restrict__ W2,
                           __bf16* __restrict__ w1t, __bf16* __restrict__ w2t) {
    int idx = blockIdx.x * blockDim.x + threadIdx.x;
    if (idx < 4 * IN1 * GH) {             // W1 [4][1024][32] -> w1t [4][32][1024]
        int g = idx / (IN1 * GH);
        int r = idx - g * (IN1 * GH);
        int k = r / GH;
        int n = r - k * GH;
        w1t[(g * GH + n) * IN1 + k] = (__bf16)W1[idx];
    }
    if (idx < 4 * GH * H_DIM) {           // W2 [4][32][512] -> w2t [4][512][32]
        int g = idx / (GH * H_DIM);
        int r = idx - g * (GH * H_DIM);
        int k = r / H_DIM;
        int n = r - k * H_DIM;
        w2t[(g * H_DIM + n) * GH + k] = (__bf16)W2[idx];
    }
}

// ---- persistent recurrent kernel: one workgroup owns BT batch rows for all T ----
__global__ void __launch_bounds__(256) qlstm_main(
    const float* __restrict__ x,            // [T,B,D]
    const float* __restrict__ hx,           // [B,H]
    const float* __restrict__ cx,           // [B,H]
    const float* __restrict__ b1,           // [4,GH]
    const float* __restrict__ b2,           // [4,H]
    const __bf16* __restrict__ w1t,         // [4][GH][IN1] bf16
    const __bf16* __restrict__ w2t,         // [4][H][GH]   bf16
    float* __restrict__ out)                // outputs[T,B,H] ++ hf[B,H] ++ cf[B,H]
{
    extern __shared__ char smem[];
    __bf16* smW1 = (__bf16*)smem;                        // 4*32*1024  (256 KB)
    __bf16* comb = (__bf16*)(smem + 262144);             // 16*1024    (32 KB)
    __bf16* z1   = (__bf16*)(smem + 262144 + 32768);     // 4*16*32    (4 KB)

    const int tid   = threadIdx.x;
    const int wave  = tid >> 5;
    const int lane  = tid & 31;
    const int lhalf = lane >> 4;     // 0: lanes 0-15, 1: lanes 16-31
    const int l16   = lane & 15;
    const int bbase = blockIdx.x * BT;

    float* hf = out + (size_t)T_DIM * B_DIM * H_DIM;
    float* cf = hf + (size_t)B_DIM * H_DIM;

    // ---- one-time: W1t (bf16) global -> LDS ----
    {
        const v4u* src = (const v4u*)w1t;
        v4u*       dst = (v4u*)smW1;
        for (int i = tid; i < (4 * GH * IN1) / 8; i += 256) dst[i] = src[i];
    }
    // ---- one-time: h0 -> comb h-part (bf16) ----
    for (int i = tid; i < BT * H_DIM; i += 256) {
        int row = i >> 9, col = i & 511;
        comb[row * IN1 + 512 + col] = (__bf16)hx[(bbase + row) * H_DIM + col];
    }

    // ---- per-wave constants ----
    const int   g1  = wave >> 1;           // layer-1: gate owned by this wave
    const int   ntz = wave & 1;            // layer-1: which 16-col half of GH
    const float b1v = b1[g1 * GH + ntz * 16 + l16];

    float b2v[4][4];
    v8f   cacc[4];                          // persistent c-state, never leaves VGPRs
    #pragma unroll
    for (int j = 0; j < 4; ++j) {
        int nt = wave * 4 + j;
        int hcol = nt * 16 + l16;
        #pragma unroll
        for (int g = 0; g < 4; ++g) b2v[j][g] = b2[g * H_DIM + hcol];
        #pragma unroll
        for (int r = 0; r < 8; ++r)
            cacc[j][r] = cx[(bbase + r + 8 * lhalf) * H_DIM + hcol];
    }

    // per-lane base pointers (M/N = l16; k = 16*half + s, same order for A and B)
    const __bf16* aRow  = comb + l16 * IN1;                 // A: M = l16
    const int ka = lhalf ? 16 : 0;                          // A: K sub-block
    const int kb = lhalf ? 16 : 0;                          // B: K sub-block
    const __bf16* bRow1 = smW1 + (g1 * GH + ntz * 16 + l16) * IN1;  // B: N = col
    const __bf16* zRowA = z1 + l16 * GH + kb;               // layer-2 A base (per gate: +g*16*GH)
    // layer-2 B base for this wave: tiles nt = wave*4 + j live at fixed offsets from here
    const __bf16* w2Row = w2t + ((size_t)(wave * 4 * 16 + l16)) * GH + kb;
    __bf16* zStore = z1 + (g1 * 16) * GH + ntz * 16 + l16;

    __syncthreads();

    for (int t = 0; t < T_DIM; ++t) {
        // ---- stage x[t, btile, :] f32 -> bf16 into comb ----
        const float* xt = x + ((size_t)t * B_DIM + bbase) * D_DIM;  // contiguous 16*512 floats
        for (int i = tid; i < (BT * D_DIM) / 4; i += 256) {
            v4f vx = ((const v4f*)xt)[i];
            int e   = i * 4;
            int row = e >> 9, col = e & 511;
            v4bf pk = { (__bf16)vx.x, (__bf16)vx.y, (__bf16)vx.z, (__bf16)vx.w };
            *(v4bf*)(comb + row * IN1 + col) = pk;
        }
        if (t + 1 < T_DIM) {
            const float* xn = x + ((size_t)(t + 1) * B_DIM + bbase) * D_DIM;
            __builtin_prefetch(xn + tid * 32, 0, 0);   // cover next 32KB tile
        }
        __syncthreads();

        // ---- layer 1: z1[g1][:, ntz half] = relu(comb @ W1t[g1] + b1), K = 1024 ----
        // two independent accumulator chains to keep the matrix pipe busy
        v8f acc0 = {}, acc1 = {};
        #pragma unroll 2
        for (int kt = 0; kt < IN1 / 32; kt += 2) {
            union { v16bf v; v8bf h[2]; } A0, A1;
            A0.h[0] = *(const v8bf*)(aRow + kt * 32 + ka);
            A0.h[1] = *(const v8bf*)(aRow + kt * 32 + ka + 8);
            v16bf B0 = *(const v16bf*)(bRow1 + kt * 32 + kb);
            A1.h[0] = *(const v8bf*)(aRow + (kt + 1) * 32 + ka);
            A1.h[1] = *(const v8bf*)(aRow + (kt + 1) * 32 + ka + 8);
            v16bf B1 = *(const v16bf*)(bRow1 + (kt + 1) * 32 + kb);
            acc0 = __builtin_amdgcn_wmma_f32_16x16x32_bf16(
                       false, A0.v, false, B0, (short)0, acc0, false, false);
            acc1 = __builtin_amdgcn_wmma_f32_16x16x32_bf16(
                       false, A1.v, false, B1, (short)0, acc1, false, false);
        }
        {   // combine chains + bias + relu, store z1 as bf16 (A-operand of layer 2)
            #pragma unroll
            for (int r = 0; r < 8; ++r) {
                int row = r + 8 * lhalf;                 // C layout: M = r + 8*half
                float v = acc0[r] + acc1[r] + b1v;
                zStore[row * GH] = (__bf16)(v > 0.0f ? v : 0.0f);
            }
        }
        __syncthreads();

        // ---- layer 2 (all 4 gates per owned tile) + LSTM cell update ----
        v16bf A2[4];                         // z1 A-tiles: reused across all 4 owned n-tiles
        #pragma unroll
        for (int g = 0; g < 4; ++g) A2[g] = *(const v16bf*)(zRowA + g * 16 * GH);

        #pragma unroll
        for (int j = 0; j < 4; ++j) {
            int nt   = wave * 4 + j;
            int hcol = nt * 16 + l16;
            v8f gaccs[4];
            #pragma unroll
            for (int g = 0; g < 4; ++g) {
                v16bf B2 = *(const v16bf*)(w2Row + (g * H_DIM + j * 16) * GH);
                v8f zero = {};
                gaccs[g] = __builtin_amdgcn_wmma_f32_16x16x32_bf16(
                               false, A2[g], false, B2, (short)0, zero, false, false);
            }
            v8f cs = cacc[j];
            #pragma unroll
            for (int r = 0; r < 8; ++r) {
                float fg = sigmoid_fast(gaccs[0][r] + b2v[j][0]);
                float ig = sigmoid_fast(gaccs[1][r] + b2v[j][1]);
                float gg = tanh_fast   (gaccs[2][r] + b2v[j][2]);
                float og = sigmoid_fast(gaccs[3][r] + b2v[j][3]);
                float cn = fg * cs[r] + ig * gg;
                cs[r] = cn;
                float hn = og * tanh_fast(cn);
                int row = r + 8 * lhalf;
                int b   = bbase + row;
                out[((size_t)t * B_DIM + b) * H_DIM + hcol] = hn;     // outputs[t]
                comb[row * IN1 + 512 + hcol] = (__bf16)hn;            // h for next step
                if (t == T_DIM - 1) hf[(size_t)b * H_DIM + hcol] = hn;
            }
            cacc[j] = cs;
        }
        __syncthreads();
    }

    // ---- final cell state ----
    #pragma unroll
    for (int j = 0; j < 4; ++j) {
        int nt = wave * 4 + j;
        int hcol = nt * 16 + l16;
        #pragma unroll
        for (int r = 0; r < 8; ++r) {
            int b = bbase + r + 8 * lhalf;
            cf[(size_t)b * H_DIM + hcol] = cacc[j][r];
        }
    }
}

extern "C" void kernel_launch(void* const* d_in, const int* in_sizes, int n_in,
                              void* d_out, int out_size, void* d_ws, size_t ws_size,
                              hipStream_t stream) {
    const float* x  = (const float*)d_in[0];
    const float* hx = (const float*)d_in[1];
    const float* cx = (const float*)d_in[2];
    const float* W1 = (const float*)d_in[3];
    const float* b1 = (const float*)d_in[4];
    const float* W2 = (const float*)d_in[5];
    const float* b2 = (const float*)d_in[6];

    __bf16* w1t = (__bf16*)d_ws;                               // 256 KB
    __bf16* w2t = (__bf16*)((char*)d_ws + (size_t)4*GH*IN1*2); // 128 KB
    float* out = (float*)d_out;

    qlstm_prep<<<(4 * IN1 * GH + 255) / 256, 256, 0, stream>>>(W1, W2, w1t, w2t);

    size_t lds_bytes = 262144 + 32768 + 4096;   // W1t + comb + z1 = 292 KB (<= 320 KB/WGP)
    qlstm_main<<<NWG, 256, lds_bytes, stream>>>(x, hx, cx, b1, b2, w1t, w2t, out);
}
